// WaveNetNeuro_55276229100101
// MI455X (gfx1250) — compile-verified, matrix operand order checked by reading, not run on hardware
//
#include <hip/hip_runtime.h>
#include <hip/hip_bf16.h>

typedef __bf16 bf16_t;
typedef __attribute__((ext_vector_type(16))) __bf16 v16bf;
typedef __attribute__((ext_vector_type(8)))  float  v8f;

#define B_    128
#define S_    256
#define E_    256
#define C_    256
#define C2_   512
#define HH    16
#define WW    16
#define NC_   2
#define MAXST 30
#define DT_   0.3f
#define THR_  0.1f
#define EPS_  1e-5f

__device__ __forceinline__ float gelu_exact(float x) {
  return 0.5f * x * (1.0f + erff(x * 0.70710678118654752f));
}

// ---------------------------------------------------------------------------
// Fragment element mapping (wave32 WMMA bf16 16x16x32):
//   lane = khalf*16 + r   (khalf = lane>>4, r = lane&15)
//   A frag (16xK): M = m0 + r,  K(i) = k0 + (i>>3)*16 + khalf*8 + (i&7)
//   B frag (Kx16): N = n0 + r,  K(i) = same formula
//   C/D frag f32:  N = r, M = khalf*8 + v  (v = element 0..7)
// Packed buffers store fragment f, lane l at offset (f*32 + l)*16 bf16
// (32 B per lane -> one coalesced 1 KB burst per wave per fragment).
// ---------------------------------------------------------------------------

// Pack to_field_w (B-frags), r1_w (A-frags), r2_w (A-frags) into bf16 frag order.
__global__ void pack_weights_kernel(const float* __restrict__ tfw,
                                    const float* __restrict__ r1w,
                                    const float* __restrict__ r2w,
                                    bf16_t* __restrict__ ptf,
                                    bf16_t* __restrict__ pr1,
                                    bf16_t* __restrict__ pr2) {
  int gid = blockIdx.x * blockDim.x + threadIdx.x;
  const int N0 = 8 * 16 * 32 * 16;   // to_field: frag = kt*16+nt
  const int N1 = 32 * 8 * 32 * 16;   // r1:       frag = mt*8+kt
  const int N2 = 16 * 16 * 32 * 16;  // r2:       frag = mt*16+kt
  if (gid < N0) {
    int i = gid & 15, lane = (gid >> 4) & 31, frag = gid >> 9;
    int kt = frag >> 4, nt = frag & 15;
    int khalf = lane >> 4, r = lane & 15;
    int kk = kt * 32 + ((i >> 3) * 16) + khalf * 8 + (i & 7);
    ptf[gid] = (bf16_t)tfw[kk * C_ + nt * 16 + r];
  } else if (gid < N0 + N1) {
    int e = gid - N0;
    int i = e & 15, lane = (e >> 4) & 31, frag = e >> 9;
    int mt = frag >> 3, kt = frag & 7;
    int khalf = lane >> 4, r = lane & 15;
    int kk = kt * 32 + ((i >> 3) * 16) + khalf * 8 + (i & 7);
    pr1[e] = (bf16_t)r1w[(mt * 16 + r) * C_ + kk];
  } else if (gid < N0 + N1 + N2) {
    int e = gid - N0 - N1;
    int i = e & 15, lane = (e >> 4) & 31, frag = e >> 9;
    int mt = frag >> 4, kt = frag & 15;
    int khalf = lane >> 4, r = lane & 15;
    int kk = kt * 32 + ((i >> 3) * 16) + khalf * 8 + (i & 7);
    pr2[e] = (bf16_t)r2w[(mt * 16 + r) * C2_ + kk];
  }
}

// Embedding gather + pos add + projection to field[b][c][s] via bf16 WMMA.
// One wave per 16-row s-tile; A frags (emb rows) built on the fly.
__global__ void embed_project_kernel(const int* __restrict__ x,
                                     const float* __restrict__ emb,
                                     const float* __restrict__ pos,
                                     const bf16_t* __restrict__ ptf,
                                     const float* __restrict__ tfb,
                                     float* __restrict__ field) {
  int lane = threadIdx.x & 31;
  int wave = threadIdx.x >> 5;
  int tile = blockIdx.x * 4 + wave;         // 2048 = B * (S/16)
  int b = tile >> 4, st = tile & 15;
  int s0 = st * 16;
  int khalf = lane >> 4, r = lane & 15;
  int srow = s0 + r;
  int tok = x[b * S_ + srow];
  const float* erow = emb + (long)tok * E_;
  const float* prow = pos + (long)srow * E_;

  v16bf a[8];
#pragma unroll
  for (int kt = 0; kt < 8; ++kt) {
#pragma unroll
    for (int i = 0; i < 16; ++i) {
      int kk = kt * 32 + ((i >> 3) * 16) + khalf * 8 + (i & 7);
      a[kt][i] = (bf16_t)(erow[kk] + prow[kk]);
    }
  }

  const bf16_t* pbl = ptf + lane * 16;      // per-lane fragment base
  for (int ct = 0; ct < 16; ++ct) {
    v8f acc = {0.f, 0.f, 0.f, 0.f, 0.f, 0.f, 0.f, 0.f};
#pragma unroll
    for (int kt = 0; kt < 8; ++kt) {
      v16bf bw = *(const v16bf*)(pbl + (kt * 16 + ct) * 512);
      acc = __builtin_amdgcn_wmma_f32_16x16x32_bf16(false, a[kt], false, bw,
                                                    (short)0, acc, false, false);
    }
    int c = ct * 16 + r;
    float bias = tfb[c];
    float* dst = field + (((long)b * C_ + c) * S_) + s0 + khalf * 8;
#pragma unroll
    for (int v = 0; v < 8; ++v) dst[v] = acc[v] + bias;
  }
}

// Persistent per-example evolution: one workgroup (8 waves) per batch example,
// all 30 steps inside the kernel.  GEMMs in bf16 WMMA, everything else f32.
__global__ void evolve_kernel(const float* __restrict__ dw,
                              const float* __restrict__ db,
                              const float* __restrict__ r1b,
                              const float* __restrict__ r2b,
                              const float* __restrict__ coeffp,
                              const float* __restrict__ nw,
                              const float* __restrict__ nb,
                              const float* __restrict__ h1w,
                              const float* __restrict__ h1b,
                              const float* __restrict__ h2w,
                              const float* __restrict__ h2b,
                              const bf16_t* __restrict__ pr1,
                              const bf16_t* __restrict__ pr2,
                              float* __restrict__ fieldAll,
                              float* __restrict__ tAll,
                              float* __restrict__ finAll,
                              float* __restrict__ out) {
  extern __shared__ char smem[];
  bf16_t* hf   = (bf16_t*)smem;                 // 8 waves * 16 frags * 32 lanes * 16 bf16 = 128KB
  float* red1  = (float*)(smem + 131072);       // 256 f32
  float* red2  = red1 + 256;                    // 256 f32
  float* pooled= red2 + 256;                    // 256 f32
  float* hid   = pooled + 256;                  // 128 f32
  float* ctrl  = hid + 128;                     // [0]=init_chg [1]=steps
  int*   ictrl = (int*)(ctrl + 2);              // [0]=converged [1]=newly

  int tid = threadIdx.x;
  int lane = tid & 31, wave = tid >> 5;
  int khalf = lane >> 4, r = lane & 15;
  int b = blockIdx.x;
  float* F   = fieldAll + (long)b * (C_ * S_);
  float* T   = tAll     + (long)b * (C_ * S_);
  float* FIN = finAll   + (long)b * (C_ * S_);
  float coeff = coeffp[0];

  if (tid == 0) { ctrl[0] = 1.0f; ctrl[1] = (float)MAXST; ictrl[0] = 0; ictrl[1] = 0; }
  __syncthreads();

  const bf16_t* pa1l = pr1 + lane * 16;   // per-lane fragment bases
  const bf16_t* pa2l = pr2 + lane * 16;

  for (int step = 0; step < MAXST; ++step) {
    float lsum = 0.f, lsq = 0.f;

    // ---- phase 1: t = coeff*(conv3x3(f)+db) + r2w @ gelu(r1w @ f + r1b) + r2b
    for (int ct = wave; ct < 16; ct += 8) {
      int s0 = ct * 16;
      // B fragments of f (bf16 from f32 field in L2)
      v16bf bf[8];
#pragma unroll
      for (int kt = 0; kt < 8; ++kt) {
#pragma unroll
        for (int i = 0; i < 16; ++i) {
          int kk = kt * 32 + ((i >> 3) * 16) + khalf * 8 + (i & 7);
          bf[kt][i] = (bf16_t)F[kk * S_ + s0 + r];
        }
      }
      // r1 GEMM: h tiles -> gelu -> bf16 frags in per-wave LDS (r2 B-operand layout)
      for (int mt = 0; mt < 32; ++mt) {
        __builtin_prefetch(pa1l + (((mt + 1) & 31) * 8) * 512, 0, 1);
        v8f acc = {0.f, 0.f, 0.f, 0.f, 0.f, 0.f, 0.f, 0.f};
#pragma unroll
        for (int kt = 0; kt < 8; ++kt) {
          v16bf aw = *(const v16bf*)(pa1l + (mt * 8 + kt) * 512);
          acc = __builtin_amdgcn_wmma_f32_16x16x32_bf16(false, aw, false, bf[kt],
                                                        (short)0, acc, false, false);
        }
        int j = mt >> 1;
        bf16_t* dst = hf + (((wave * 16 + j) * 32 + lane) * 16) + ((mt & 1) * 8);
#pragma unroll
        for (int v = 0; v < 8; ++v) {
          int m = mt * 16 + khalf * 8 + v;
          dst[v] = (bf16_t)gelu_exact(acc[v] + r1b[m]);
        }
      }
      // r2 GEMM + diffusion + bias -> t
      for (int mt2 = 0; mt2 < 16; ++mt2) {
        __builtin_prefetch(pa2l + (((mt2 + 1) & 15) * 16) * 512, 0, 1);
        v8f acc = {0.f, 0.f, 0.f, 0.f, 0.f, 0.f, 0.f, 0.f};
#pragma unroll
        for (int j = 0; j < 16; ++j) {
          v16bf aw = *(const v16bf*)(pa2l + (mt2 * 16 + j) * 512);
          v16bf bh = *(const v16bf*)(hf + (((wave * 16 + j) * 32 + lane) * 16));
          acc = __builtin_amdgcn_wmma_f32_16x16x32_bf16(false, aw, false, bh,
                                                        (short)0, acc, false, false);
        }
#pragma unroll
        for (int v = 0; v < 8; ++v) {
          int c = mt2 * 16 + khalf * 8 + v;
          const float* fr = F + c * S_;
          const float* w9 = dw + c * 9;
          float d = 0.f;
#pragma unroll
          for (int ky = 0; ky < 3; ++ky) {
            int y = ct + ky - 1;
            if (y < 0 || y >= HH) continue;
#pragma unroll
            for (int kx = 0; kx < 3; ++kx) {
              int xx = r + kx - 1;
              if (xx < 0 || xx >= WW) continue;
              d += w9[ky * 3 + kx] * fr[y * WW + xx];
            }
          }
          float t = coeff * (d + db[c]) + acc[v] + r2b[c];
          T[c * S_ + s0 + r] = t;
          lsum += t;
          lsq  += t * t;
        }
      }
    }

    // ---- GroupNorm statistics (block reduce over 65536 elements)
    __threadfence_block();
    red1[tid] = lsum; red2[tid] = lsq;
    __syncthreads();
    for (int off = 128; off > 0; off >>= 1) {
      if (tid < off) { red1[tid] += red1[tid + off]; red2[tid] += red2[tid + off]; }
      __syncthreads();
    }
    float mu  = red1[0] * (1.f / 65536.f);
    float var = red2[0] * (1.f / 65536.f) - mu * mu;
    float inv = rsqrtf(var + EPS_);
    __syncthreads();

    // ---- phase 2: normalize, Euler step, change accumulation
    float lchg = 0.f;
    for (int e2 = 0; e2 < 256; ++e2) {
      int off = e2 * 256 + tid;            // c = e2, s = tid (coalesced)
      float t  = T[off];
      float tn = (t - mu) * inv * nw[e2] + nb[e2];
      float f  = F[off];
      float nf = f + DT_ * (tn - f);
      lchg += fabsf(nf - f);
      F[off] = nf;
    }
    __threadfence_block();
    red1[tid] = lchg;
    __syncthreads();
    for (int off = 128; off > 0; off >>= 1) {
      if (tid < off) red1[tid] += red1[tid + off];
      __syncthreads();
    }
    if (tid == 0) {
      float chg = red1[0] * (1.f / 65536.f);
      if (step == 0) ctrl[0] = fmaxf(chg, 1e-8f);
      float rel = chg / ctrl[0];
      if (!ictrl[0] && rel < THR_) { ictrl[0] = 1; ictrl[1] = 1; ctrl[1] = (float)(step + 1); }
    }
    __syncthreads();
    if (ictrl[1]) {  // snapshot field at first convergence
      for (int o = tid; o < C_ * S_; o += 256) FIN[o] = F[o];
    }
    __syncthreads();
    if (tid == 0) ictrl[1] = 0;
    __syncthreads();
  }

  if (!ictrl[0]) {  // never converged: final = last field
    for (int o = tid; o < C_ * S_; o += 256) FIN[o] = F[o];
  }
  __threadfence_block();
  __syncthreads();

  // ---- pooled mean over spatial dims, then tiny head GEMMs
  {
    float s = 0.f;
    const float* fr = FIN + tid * S_;
    for (int i = 0; i < S_; ++i) s += fr[i];
    pooled[tid] = s * (1.f / 256.f);
  }
  __syncthreads();
  if (tid < 128) {
    float a = h1b[tid];
    for (int c = 0; c < 256; ++c) a += pooled[c] * h1w[c * 128 + tid];
    hid[tid] = gelu_exact(a);
  }
  __syncthreads();
  if (tid < NC_) {
    float a = h2b[tid];
    for (int j = 0; j < 128; ++j) a += hid[j] * h2w[j * NC_ + tid];
    out[b * NC_ + tid] = a;
  }
  if (tid == 0) out[B_ * NC_ + b] = ctrl[1];   // steps (f32), after the logits
}

extern "C" void kernel_launch(void* const* d_in, const int* in_sizes, int n_in,
                              void* d_out, int out_size, void* d_ws, size_t ws_size,
                              hipStream_t stream) {
  (void)in_sizes; (void)n_in; (void)out_size; (void)ws_size;
  const int*   x     = (const int*)  d_in[0];
  const float* emb   = (const float*)d_in[1];
  const float* pos   = (const float*)d_in[2];
  const float* tfw   = (const float*)d_in[3];
  const float* tfb   = (const float*)d_in[4];
  const float* dw    = (const float*)d_in[5];
  const float* db    = (const float*)d_in[6];
  const float* r1w   = (const float*)d_in[7];
  const float* r1b   = (const float*)d_in[8];
  const float* r2w   = (const float*)d_in[9];
  const float* r2b   = (const float*)d_in[10];
  const float* coeff = (const float*)d_in[11];
  const float* nw    = (const float*)d_in[12];
  const float* nb    = (const float*)d_in[13];
  const float* h1w   = (const float*)d_in[14];
  const float* h1b   = (const float*)d_in[15];
  const float* h2w   = (const float*)d_in[16];
  const float* h2b   = (const float*)d_in[17];
  float* out = (float*)d_out;

  // Workspace layout (needs ~96.7 MB; everything L2-resident on MI455X):
  char* ws = (char*)d_ws;
  float*  field = (float*)(ws + 0);            // 33,554,432 B : f32 field [B][C][S]
  float*  tbuf  = (float*)(ws + 33554432);     // 33,554,432 B : f32 t     [B][C][S]
  float*  fin   = (float*)(ws + 67108864);     // 33,554,432 B : f32 final [B][C][S]
  bf16_t* ptf   = (bf16_t*)(ws + 100663296);   //    131,072 B : to_field_w B-frags
  bf16_t* pr1   = (bf16_t*)(ws + 100794368);   //    262,144 B : r1_w A-frags
  bf16_t* pr2   = (bf16_t*)(ws + 101056512);   //    262,144 B : r2_w A-frags

  pack_weights_kernel<<<1280, 256, 0, stream>>>(tfw, r1w, r2w, ptf, pr1, pr2);
  embed_project_kernel<<<512, 128, 0, stream>>>(x, emb, pos, ptf, tfb, field);
  evolve_kernel<<<128, 256, 135168, stream>>>(dw, db, r1b, r2b, coeff, nw, nb,
                                              h1w, h1b, h2w, h2b, pr1, pr2,
                                              field, tbuf, fin, out);
}
